// FakeMoE_41274635714717
// MI455X (gfx1250) — matrix operand order repaired and checked by reading, not compiled, not run on hardware
//
#include <hip/hip_runtime.h>
#include <hip/hip_bf16.h>

#define HID   1024
#define FFN   2816
#define NEXP  8
#define NTOK  2048
#define MAXROWS 4352   // 4096 assignments + padding + 64-row supertile overhang

typedef __attribute__((ext_vector_type(16))) __bf16 v16bf;
typedef __attribute__((ext_vector_type(8)))  float  v8f;

// ---------- helpers ----------
static __device__ __forceinline__ unsigned short f2bf_bits(float f) {
    unsigned u = __builtin_bit_cast(unsigned, f);
    u += 0x7FFFu + ((u >> 16) & 1u);          // round-to-nearest-even
    return (unsigned short)(u >> 16);
}
static __device__ __forceinline__ __bf16 f2bf(float f) {
    unsigned short h = f2bf_bits(f);
    return __builtin_bit_cast(__bf16, h);
}

// Pack two fp32 into packed bf16 pair (truncation) with a single v_perm_b32.
static __device__ __forceinline__ unsigned pack2bf(float lo, float hi) {
    return __builtin_amdgcn_perm(__builtin_bit_cast(unsigned, hi),
                                 __builtin_bit_cast(unsigned, lo),
                                 0x07060302u);
}

// Load 16 consecutive fp32, truncate-convert to a bf16 WMMA B fragment (8 v_perm total).
static __device__ __forceinline__ v16bf load16f_bf(const float* __restrict__ p) {
    const float4* q = (const float4*)p;
    float4 a = q[0], b = q[1], c = q[2], d = q[3];
    union { v16bf v; unsigned u[8]; } r;
    r.u[0] = pack2bf(a.x, a.y); r.u[1] = pack2bf(a.z, a.w);
    r.u[2] = pack2bf(b.x, b.y); r.u[3] = pack2bf(b.z, b.w);
    r.u[4] = pack2bf(c.x, c.y); r.u[5] = pack2bf(c.z, c.w);
    r.u[6] = pack2bf(d.x, d.y); r.u[7] = pack2bf(d.z, d.w);
    return r.v;
}

// WMMA A slice from a bf16 row (K-major), unconditional.
// lanes 0-15: K = k0..k0+15 ; lanes 16-31: K = k0+16..k0+31
static __device__ __forceinline__ v16bf load_a_bf(const __bf16* __restrict__ row,
                                                  int k0, int hf) {
    union U { v16bf v; uint4 q[2]; } u;
    u.q[0] = *(const uint4*)(row + k0 + hf * 16);
    u.q[1] = *(const uint4*)(row + k0 + hf * 16 + 8);
    return u.v;
}

static __device__ __forceinline__ v8f wmma_bf16(v16bf a, v16bf b, v8f c) {
    return __builtin_amdgcn_wmma_f32_16x16x32_bf16(false, a, false, b, (short)0, c, false, false);
}

// silu via hardware v_rcp_f32 (1-ulp; far below bf16 noise floor)
static __device__ __forceinline__ float silu_fast(float g) {
    return g * __builtin_amdgcn_rcpf(1.0f + __expf(-g));
}

// ---------- kernel 1: zero output + counters + dummy zero row ----------
__global__ __launch_bounds__(256) void moe_zero_kernel(float* __restrict__ out, int n,
                                                       int* __restrict__ counters,
                                                       unsigned* __restrict__ zrow) {
    int gid = blockIdx.x * 256 + threadIdx.x;
    if (gid < n) out[gid] = 0.0f;
    if (gid < 16) counters[gid] = 0;   // [0..7]=count, [8..15]=cursor
    if (gid < FFN / 2) zrow[gid] = 0u; // FFN bf16 of zeros
}

// ---------- kernel 2: fp32 -> bf16 activation convert (RNE, done once) ----------
__global__ __launch_bounds__(256) void moe_cvt_kernel(const float* __restrict__ x,
                                                      __bf16* __restrict__ xb, int n4) {
    int gid = blockIdx.x * 256 + threadIdx.x;
    if (gid >= n4) return;
    float4 v = ((const float4*)x)[gid];
    uint2 o;
    o.x = (unsigned)f2bf_bits(v.x) | ((unsigned)f2bf_bits(v.y) << 16);
    o.y = (unsigned)f2bf_bits(v.z) | ((unsigned)f2bf_bits(v.w) << 16);
    ((uint2*)xb)[gid] = o;
}

// ---------- kernel 3: router (one wave32 per token) ----------
__global__ __launch_bounds__(256) void moe_router_kernel(const float* __restrict__ x,
                                                         const float* __restrict__ Wgate,
                                                         int* __restrict__ counters,
                                                         int* __restrict__ topki,
                                                         float* __restrict__ topkw) {
    int wave = threadIdx.x >> 5, lane = threadIdx.x & 31;
    int t = blockIdx.x * 8 + wave;
    if (t >= NTOK) return;
    const float* xr = x + (size_t)t * HID;
    float acc[NEXP];
#pragma unroll
    for (int e = 0; e < NEXP; e++) acc[e] = 0.0f;
    for (int k = lane; k < HID; k += 32) {
        float xv = xr[k];
#pragma unroll
        for (int e = 0; e < NEXP; e++) acc[e] = fmaf(xv, Wgate[e * HID + k], acc[e]);
    }
#pragma unroll
    for (int e = 0; e < NEXP; e++) {
        for (int off = 16; off >= 1; off >>= 1)
            acc[e] += __shfl_xor(acc[e], off, 32);
    }
    if (lane == 0) {
        float mx = acc[0];
#pragma unroll
        for (int e = 1; e < NEXP; e++) mx = fmaxf(mx, acc[e]);
        float ex[NEXP];
#pragma unroll
        for (int e = 0; e < NEXP; e++) ex[e] = __expf(acc[e] - mx);
        int i0 = 0;
#pragma unroll
        for (int e = 1; e < NEXP; e++) if (ex[e] > ex[i0]) i0 = e;
        int i1 = (i0 == 0) ? 1 : 0;
#pragma unroll
        for (int e = 0; e < NEXP; e++) if (e != i0 && ex[e] > ex[i1]) i1 = e;
        float rs = __builtin_amdgcn_rcpf(ex[i0] + ex[i1]);
        topki[t * 2 + 0] = i0;  topkw[t * 2 + 0] = ex[i0] * rs;
        topki[t * 2 + 1] = i1;  topkw[t * 2 + 1] = ex[i1] * rs;
        atomicAdd(&counters[i0], 1);
        atomicAdd(&counters[i1], 1);
    }
}

// ---------- kernel 4: 16-padded segment offsets ----------
__global__ void moe_offsets_kernel(int* __restrict__ counters) {
    if (threadIdx.x == 0 && blockIdx.x == 0) {
        int run = 0;
        for (int e = 0; e < NEXP; e++) {
            counters[16 + e] = run;                     // seg_off
            run += (counters[e] + 15) & ~15;
        }
    }
}

// ---------- kernel 5: scatter tokens into expert slots ----------
__global__ __launch_bounds__(256) void moe_scatter_kernel(const int* __restrict__ topki,
                                                          const float* __restrict__ topkw,
                                                          int* __restrict__ counters,
                                                          int* __restrict__ tok,
                                                          float* __restrict__ wt) {
    int t = blockIdx.x * 256 + threadIdx.x;
    if (t >= NTOK) return;
#pragma unroll
    for (int j = 0; j < 2; j++) {
        int e = topki[t * 2 + j];
        int pos = atomicAdd(&counters[8 + e], 1);       // cursor
        int slot = counters[16 + e] + pos;              // seg_off
        tok[slot] = t;
        wt[slot] = topkw[t * 2 + j];
    }
}

// ---------- kernel 6: fused gate+up GEMM + SwiGLU (bf16 WMMA) ----------
// grid: (22 n-chunks of 128, 32 m-supertiles of 64 rows, 8 experts)
// block 256 = 8 waves; wave tile = 64 rows x 16 cols, gate+up fused (8 WMMA / K-step)
__global__ __launch_bounds__(256) void moe_gateup_kernel(const float* __restrict__ Wg,
                                                         const float* __restrict__ Wu,
                                                         const __bf16* __restrict__ xb,
                                                         const int* __restrict__ counters,
                                                         const int* __restrict__ tok,
                                                         __bf16* __restrict__ hws,
                                                         const __bf16* __restrict__ zrow) {
    int e = blockIdx.z, mt = blockIdx.y;
    int c = counters[e];
    if (mt * 64 >= c) return;
    int wave = threadIdx.x >> 5, lane = threadIdx.x & 31;
    int nb = blockIdx.x * 128 + wave * 16;              // 22*128 == 2816 exactly
    int hf = lane >> 4, m = lane & 15;
    int seg = counters[16 + e];

    // invalid rows read the zero row -> straight-line inner loop, no exec branches
    const __bf16* arow[4];
#pragma unroll
    for (int st = 0; st < 4; st++) {
        int grow = mt * 64 + st * 16 + m;
        arow[st] = (grow < c) ? (xb + (size_t)tok[seg + grow] * HID) : zrow;
    }
    const float* wgp = Wg + ((size_t)e * FFN + nb + m) * HID + hf * 16;
    const float* wup = Wu + ((size_t)e * FFN + nb + m) * HID + hf * 16;

    v8f cg[4] = {};
    v8f cu[4] = {};
#pragma unroll 1
    for (int k0 = 0; k0 < HID; k0 += 32) {
        v16bf bg = load16f_bf(wgp + k0);                // 1 conversion feeds 4 WMMAs
        v16bf bu = load16f_bf(wup + k0);
#pragma unroll
        for (int st = 0; st < 4; st++) {
            v16bf a = load_a_bf(arow[st], k0, hf);
            cg[st] = wmma_bf16(a, bg, cg[st]);
            cu[st] = wmma_bf16(a, bu, cu[st]);
        }
    }
    int col = nb + m;
#pragma unroll
    for (int st = 0; st < 4; st++) {
        if (mt * 64 + st * 16 >= c) break;              // subtile past padded segment
        int rowbase = seg + mt * 64 + st * 16;
#pragma unroll
        for (int i = 0; i < 8; i++) {
            int r = i + hf * 8;
            float hv = silu_fast(cg[st][i]) * cu[st][i]; // ==0 on padded rows
            hws[(size_t)(rowbase + r) * FFN + col] = f2bf(hv);
        }
    }
}

// ---------- kernel 7: down GEMM + weighted scatter-add ----------
// grid: (8 n-chunks of 128, 32 m-supertiles, 8 experts); wave tile = 64 rows x 16 cols
__global__ __launch_bounds__(256) void moe_down_kernel(const float* __restrict__ Wd,
                                                       const __bf16* __restrict__ hws,
                                                       const int* __restrict__ counters,
                                                       const int* __restrict__ tok,
                                                       const float* __restrict__ wt,
                                                       float* __restrict__ out,
                                                       const __bf16* __restrict__ zrow) {
    int e = blockIdx.z, mt = blockIdx.y;
    int c = counters[e];
    if (mt * 64 >= c) return;
    int wave = threadIdx.x >> 5, lane = threadIdx.x & 31;
    int nb = blockIdx.x * 128 + wave * 16;              // 8*128 == 1024 exactly
    int hf = lane >> 4, m = lane & 15;
    int seg = counters[16 + e];

    const __bf16* arow[4];
#pragma unroll
    for (int st = 0; st < 4; st++) {
        int grow = mt * 64 + st * 16 + m;
        arow[st] = (grow < ((c + 15) & ~15)) ? (hws + (size_t)(seg + grow) * FFN) : zrow;
    }
    const float* wdp = Wd + ((size_t)e * HID + nb + m) * FFN + hf * 16;

    v8f acc[4] = {};
#pragma unroll 1
    for (int k0 = 0; k0 < FFN; k0 += 32) {
        v16bf b = load16f_bf(wdp + k0);                 // 1 conversion feeds 4 WMMAs
#pragma unroll
        for (int st = 0; st < 4; st++) {
            v16bf a = load_a_bf(arow[st], k0, hf);
            acc[st] = wmma_bf16(a, b, acc[st]);
        }
    }
    int col = nb + m;
#pragma unroll
    for (int st = 0; st < 4; st++) {
        if (mt * 64 + st * 16 >= c) break;
        int rowbase = seg + mt * 64 + st * 16;
#pragma unroll
        for (int i = 0; i < 8; i++) {
            int r = i + hf * 8;
            if (mt * 64 + st * 16 + r < c) {
                int t = tok[rowbase + r];
                float w = wt[rowbase + r];
                atomicAdd(out + (size_t)t * HID + col, w * acc[st][i]);
            }
        }
    }
}

// ---------- launch ----------
extern "C" void kernel_launch(void* const* d_in, const int* in_sizes, int n_in,
                              void* d_out, int out_size, void* d_ws, size_t ws_size,
                              hipStream_t stream) {
    (void)in_sizes; (void)n_in; (void)out_size; (void)ws_size;
    const float* x     = (const float*)d_in[0];   // [1,2048,1024]
    const float* Wgate = (const float*)d_in[1];   // [8,1024]
    const float* Wg    = (const float*)d_in[2];   // [8,2816,1024]
    const float* Wu    = (const float*)d_in[3];   // [8,2816,1024]
    const float* Wd    = (const float*)d_in[4];   // [8,1024,2816]
    float* out = (float*)d_out;                   // [2048,1024]

    char* ws = (char*)d_ws;
    __bf16* xb      = (__bf16*)(ws);                              // 4,194,304 B
    __bf16* hws     = (__bf16*)(ws + 4194304);                    // 4352*2816*2 = 24,510,464 B
    int*    tok     = (int*)   (ws + 28704768);                   // 17,408 B
    float*  wt      = (float*) (ws + 28722176);                   // 17,408 B
    int*    topki   = (int*)   (ws + 28739584);                   // 16,384 B
    float*  topkw   = (float*) (ws + 28755968);                   // 16,384 B
    int*    counters= (int*)   (ws + 28772352);                   // 256 B (24 used)
    __bf16* zrow    = (__bf16*)(ws + 28772608);                   // 5,632 B of zeros

    const int nOut = NTOK * HID;                  // 2,097,152
    moe_zero_kernel<<<(nOut + 255) / 256, 256, 0, stream>>>(out, nOut, counters, (unsigned*)zrow);
    moe_cvt_kernel<<<(nOut / 4 + 255) / 256, 256, 0, stream>>>(x, xb, nOut / 4);
    moe_router_kernel<<<NTOK / 8, 256, 0, stream>>>(x, Wgate, counters, topki, topkw);
    moe_offsets_kernel<<<1, 32, 0, stream>>>(counters);
    moe_scatter_kernel<<<NTOK / 256, 256, 0, stream>>>(topki, topkw, counters, tok, wt);
    moe_gateup_kernel<<<dim3(22, 32, NEXP), 256, 0, stream>>>(Wg, Wu, xb, counters, tok, hws, zrow);
    moe_down_kernel<<<dim3(8, 32, NEXP), 256, 0, stream>>>(Wd, hws, counters, tok, wt, out, zrow);
}
